// H2GCNCompact_45028437131743
// MI455X (gfx1250) — compile-verified, matrix operand order checked by reading, not run on hardware
//
#include <hip/hip_runtime.h>
#include <stdint.h>

// ---------------------------------------------------------------------------
// H2GCN forward, MI455X (gfx1250, wave32, WMMA).
//   h0 = relu(x @ W_e + b_e)                     [N,64]   (bf16 WMMA, f32 acc)
//   hA = relu([spmm1(h0) | spmm2(h0)])           [N,128]  (atomic scatter-add)
//   hB = relu([spmm1(hA) | spmm2(hA)])           [N,256]
//   out = [h0|hA|hB] @ W_c + b_c                 [N,47]   (bf16 WMMA, f32 acc)
//
// B matrices are pre-packed into per-lane WMMA fragment layout, staged into
// LDS once per block via GLOBAL_LOAD_ASYNC_TO_LDS_B128 (ASYNCcnt-tracked),
// and read with batched ds_load_b128 feeding back-to-back WMMAs.
// ---------------------------------------------------------------------------

typedef __attribute__((ext_vector_type(16))) __bf16 v16bf;
typedef __attribute__((ext_vector_type(8)))  float  v8f;
typedef __attribute__((ext_vector_type(4)))  int    i32x4;

#define IN_DIM  512
#define HIDDEN  64
#define OUT_DIM 47
#define TOT_DIM 448     // 7*HIDDEN = 14 K-tiles of 32

#define BE_BYTES (16 * 4 * 32 * 16 * 2)   // 65536: embed B fragments
#define BC_BYTES (14 * 3 * 32 * 16 * 2)   // 43008: cls   B fragments

#if __has_builtin(__builtin_amdgcn_global_load_async_to_lds_b128) && \
    __has_builtin(__builtin_amdgcn_s_wait_asynccnt)
#define USE_ASYNC_LDS 1
typedef __attribute__((address_space(1))) i32x4 gvec_t;   // global b128
typedef __attribute__((address_space(3))) i32x4 lvec_t;   // LDS b128
#endif

extern __shared__ char smem_dyn[];

// K index inside a 32-wide chunk for bf16 A/B fragments (CDNA5 ISA 7.12.2):
// fragment elem j = 2*i+p lives at K = (i&4 ? 16:0) + 8*(lane>>4) + 2*(i&3) + p
__device__ __forceinline__ int kpat(int i, int half) {
    return ((i & 4) ? 16 : 0) + (half << 3) + ((i & 3) << 1);
}

// Block-cooperative copy of packed B into LDS (16B granules), then barrier.
__device__ __forceinline__ void stage_to_lds(const char* __restrict__ g, int bytes) {
    for (int off = threadIdx.x * 16; off < bytes; off += 256 * 16) {
#ifdef USE_ASYNC_LDS
        __builtin_amdgcn_global_load_async_to_lds_b128(
            (gvec_t*)(uintptr_t)(g + off),
            (lvec_t*)(uintptr_t)(smem_dyn + off),   // low 32 bits == LDS offset
            0, 0);
#else
        *(float4*)(smem_dyn + off) = *(const float4*)(g + off);
#endif
    }
#ifdef USE_ASYNC_LDS
    __builtin_amdgcn_s_wait_asynccnt(0);
#endif
    __syncthreads();
}

__device__ __forceinline__ v16bf lds_b_frag(int frag_idx, int lane) {
    return *(const v16bf*)(smem_dyn + ((size_t)(frag_idx * 32 + lane) << 5));
}

// ---------------- pre-pack B into WMMA fragment order (bf16) ----------------
// P[((chunk*ntiles + nt)*32 + lane)*16 + j] = W[kk, nt*16 + (lane&15)]
__global__ void pack_b_kernel(const float* __restrict__ W, __bf16* __restrict__ P,
                              int kchunks, int ntiles, int valid_cols) {
    const int i = blockIdx.x * blockDim.x + threadIdx.x;
    const int total = kchunks * ntiles * 32 * 16;
    if (i >= total) return;
    const int j     = i & 15;
    const int lane  = (i >> 4) & 31;
    const int cn    = i >> 9;              // chunk*ntiles + nt
    const int nt    = cn % ntiles;
    const int chunk = cn / ntiles;
    const int half  = lane >> 4;
    const int kk    = chunk * 32 + kpat(j >> 1, half) + (j & 1);
    const int col   = nt * 16 + (lane & 15);
    const float v   = (col < valid_cols) ? W[(long long)kk * valid_cols + col] : 0.0f;
    P[i] = (__bf16)v;
}

// Build bf16 A fragment from a contiguous f32 row segment.
// a[0..7]  <- K = base + 8*half + 0..7 ; a[8..15] <- K = base+16+8*half + 0..7
__device__ __forceinline__ v16bf load_a_frag(const float* __restrict__ rowp,
                                             int base, int half) {
    const float4 p0 = *(const float4*)(rowp + base + (half << 3));
    const float4 p1 = *(const float4*)(rowp + base + (half << 3) + 4);
    const float4 p2 = *(const float4*)(rowp + base + 16 + (half << 3));
    const float4 p3 = *(const float4*)(rowp + base + 16 + (half << 3) + 4);
    v16bf a;
    a[0]=(__bf16)p0.x;  a[1]=(__bf16)p0.y;  a[2]=(__bf16)p0.z;  a[3]=(__bf16)p0.w;
    a[4]=(__bf16)p1.x;  a[5]=(__bf16)p1.y;  a[6]=(__bf16)p1.z;  a[7]=(__bf16)p1.w;
    a[8]=(__bf16)p2.x;  a[9]=(__bf16)p2.y;  a[10]=(__bf16)p2.z; a[11]=(__bf16)p2.w;
    a[12]=(__bf16)p3.x; a[13]=(__bf16)p3.y; a[14]=(__bf16)p3.z; a[15]=(__bf16)p3.w;
    return a;
}

// -------------------- embed GEMM: [N,512]x[512,64], relu+bias --------------
__global__ void embed_gemm_kernel(const float* __restrict__ X,
                                  const char* __restrict__ Bpk,
                                  const float* __restrict__ bias,
                                  float* __restrict__ H0, int n_rows) {
    stage_to_lds(Bpk, BE_BYTES);                    // all threads, before any exit

    const int lane = threadIdx.x & 31;
    const int wave = threadIdx.x >> 5;
    const int tile = blockIdx.x * 8 + wave;         // wave-uniform -> EXEC stays full
    if (tile * 16 >= n_rows) return;
    const int m    = lane & 15;
    const int half = lane >> 4;
    const float* xrow = X + ((long long)tile * 16 + m) * IN_DIM;

    v8f acc[4] = {v8f{}, v8f{}, v8f{}, v8f{}};

    for (int k0 = 0; k0 < IN_DIM; k0 += 32) {
        if (k0 + 32 < IN_DIM)
            __builtin_prefetch(xrow + k0 + 32, 0, 3);   // global_prefetch_b8
        const int chunk = k0 >> 5;
        // Batch all B-fragment LDS loads, then A, then back-to-back WMMAs.
        v16bf bfr[4];
        #pragma unroll
        for (int nt = 0; nt < 4; ++nt) bfr[nt] = lds_b_frag(chunk * 4 + nt, lane);
        const v16bf a = load_a_frag(xrow, k0, half);
        #pragma unroll
        for (int nt = 0; nt < 4; ++nt)
            acc[nt] = __builtin_amdgcn_wmma_f32_16x16x32_bf16(
                false, a, false, bfr[nt], (short)0, acc[nt], false, false);
    }
    // D layout: VGPR r, lane -> (M = r + 8*half, N = lane&15)
    #pragma unroll
    for (int nt = 0; nt < 4; ++nt) {
        const int   n  = nt * 16 + m;
        const float bv = bias[n];
        #pragma unroll
        for (int r = 0; r < 8; ++r) {
            const long long om = (long long)tile * 16 + r + 8 * half;
            H0[om * HIDDEN + n] = fmaxf(acc[nt][r] + bv, 0.0f);
        }
    }
}

// -------------------- SpMM scatter: out[dst[e]] += w[e]*H[src[e]] ----------
__global__ void spmm_scatter_kernel(const int* __restrict__ src,
                                    const int* __restrict__ dst,
                                    const float* __restrict__ w,
                                    const float* __restrict__ H,
                                    float* __restrict__ OUT,
                                    int n_edges, int dim_shift,
                                    int out_stride, int col_off) {
    const long long idx = (long long)blockIdx.x * blockDim.x + threadIdx.x;
    const int e = (int)(idx >> dim_shift);
    if (e >= n_edges) return;
    const int f = (int)(idx & ((1 << dim_shift) - 1));
    const int s = src[e];
    const int d = dst[e];
    const float v = w[e] * H[((long long)s << dim_shift) + f];
    atomicAdd(OUT + (long long)d * out_stride + col_off + f, v);
}

__global__ void relu_inplace_kernel(float* __restrict__ p, long long n) {
    const long long i = (long long)blockIdx.x * blockDim.x + threadIdx.x;
    if (i < n) p[i] = fmaxf(p[i], 0.0f);
}

// -------------------- classifier GEMM: [N,448]x[448,48->47] ----------------
__global__ void cls_gemm_kernel(const float* __restrict__ h0,
                                const float* __restrict__ hA,
                                const float* __restrict__ hB,
                                const char* __restrict__ Bpk,
                                const float* __restrict__ bias,
                                float* __restrict__ out, int n_rows) {
    stage_to_lds(Bpk, BC_BYTES);

    const int lane = threadIdx.x & 31;
    const int wave = threadIdx.x >> 5;
    const int tile = blockIdx.x * 8 + wave;
    if (tile * 16 >= n_rows) return;
    const int m    = lane & 15;
    const int half = lane >> 4;
    const long long row = (long long)tile * 16 + m;

    v8f acc[3] = {v8f{}, v8f{}, v8f{}};

    for (int k0 = 0; k0 < TOT_DIM; k0 += 32) {
        // Rep-source selection is uniform per 32-chunk (boundaries 64/192 % 32 == 0).
        const float* srcp; int stride, kb;
        if (k0 < 64)        { srcp = h0; stride = 64;  kb = k0;       }
        else if (k0 < 192)  { srcp = hA; stride = 128; kb = k0 - 64;  }
        else                { srcp = hB; stride = 256; kb = k0 - 192; }

        const int chunk = k0 >> 5;
        v16bf bfr[3];
        #pragma unroll
        for (int nt = 0; nt < 3; ++nt) bfr[nt] = lds_b_frag(chunk * 3 + nt, lane);
        const v16bf a = load_a_frag(srcp + row * stride, kb, half);
        #pragma unroll
        for (int nt = 0; nt < 3; ++nt)
            acc[nt] = __builtin_amdgcn_wmma_f32_16x16x32_bf16(
                false, a, false, bfr[nt], (short)0, acc[nt], false, false);
    }
    #pragma unroll
    for (int nt = 0; nt < 3; ++nt) {
        const int n = nt * 16 + m;
        if (n < OUT_DIM) {
            const float bv = bias[n];
            #pragma unroll
            for (int r = 0; r < 8; ++r) {
                const long long om = (long long)tile * 16 + r + 8 * half;
                out[om * OUT_DIM + n] = acc[nt][r] + bv;
            }
        }
    }
}

// ---------------------------------------------------------------------------
extern "C" void kernel_launch(void* const* d_in, const int* in_sizes, int n_in,
                              void* d_out, int out_size, void* d_ws, size_t ws_size,
                              hipStream_t stream) {
    (void)n_in; (void)out_size; (void)ws_size;

    const float* x       = (const float*)d_in[0];
    const int*   src1    = (const int*)  d_in[1];
    const int*   dst1    = (const int*)  d_in[2];
    const float* w1      = (const float*)d_in[3];
    const int*   src2    = (const int*)  d_in[4];
    const int*   dst2    = (const int*)  d_in[5];
    const float* w2      = (const float*)d_in[6];
    const float* embed_w = (const float*)d_in[7];
    const float* embed_b = (const float*)d_in[8];
    const float* cls_w   = (const float*)d_in[9];
    const float* cls_b   = (const float*)d_in[10];
    float*       out     = (float*)d_out;

    const int E1 = in_sizes[1];
    const int E2 = in_sizes[4];
    const int N  = in_sizes[0] / IN_DIM;

    // Workspace layout (all offsets 256B-aligned).
    char* ws = (char*)d_ws;
    __bf16* Be = (__bf16*)(ws);                          // 65536 B
    __bf16* Bc = (__bf16*)(ws + BE_BYTES);               // 43008 B
    float*  h0 = (float*) (ws + 108544);                 // N*64*4
    float*  hA = (float*) (ws + 108544 + (size_t)N * 64 * 4);
    float*  hB = (float*) (ws + 108544 + (size_t)N * 64 * 4 + (size_t)N * 128 * 4);

    // Zero SpMM accumulators (required every call; graph-capture safe).
    (void)hipMemsetAsync(hA, 0, (size_t)N * 128 * sizeof(float), stream);
    (void)hipMemsetAsync(hB, 0, (size_t)N * 256 * sizeof(float), stream);

    // Pre-pack weights into WMMA B-fragment order (bf16).
    pack_b_kernel<<<(16 * 4 * 32 * 16 + 255) / 256, 256, 0, stream>>>(
        embed_w, Be, 16, 4, HIDDEN);
    pack_b_kernel<<<(14 * 3 * 32 * 16 + 255) / 256, 256, 0, stream>>>(
        cls_w, Bc, 14, 3, OUT_DIM);

    // h0 = relu(x @ We + be)
    const int row_tiles = (N + 15) / 16;
    const int gemm_blocks = (row_tiles + 7) / 8;
    embed_gemm_kernel<<<gemm_blocks, 256, BE_BYTES, stream>>>(
        x, (const char*)Be, embed_b, h0, N);

    // Round 1: hA = relu([spmm1(h0) | spmm2(h0)])   (dim=64, stride=128)
    {
        const long long t1 = (long long)E1 * 64, t2 = (long long)E2 * 64;
        spmm_scatter_kernel<<<(unsigned)((t1 + 255) / 256), 256, 0, stream>>>(
            src1, dst1, w1, h0, hA, E1, 6, 128, 0);
        spmm_scatter_kernel<<<(unsigned)((t2 + 255) / 256), 256, 0, stream>>>(
            src2, dst2, w2, h0, hA, E2, 6, 128, 64);
        const long long na = (long long)N * 128;
        relu_inplace_kernel<<<(unsigned)((na + 255) / 256), 256, 0, stream>>>(hA, na);
    }

    // Round 2: hB = relu([spmm1(hA) | spmm2(hA)])   (dim=128, stride=256)
    {
        const long long t1 = (long long)E1 * 128, t2 = (long long)E2 * 128;
        spmm_scatter_kernel<<<(unsigned)((t1 + 255) / 256), 256, 0, stream>>>(
            src1, dst1, w1, hA, hB, E1, 7, 256, 0);
        spmm_scatter_kernel<<<(unsigned)((t2 + 255) / 256), 256, 0, stream>>>(
            src2, dst2, w2, hA, hB, E2, 7, 256, 128);
        const long long nb = (long long)N * 256;
        relu_inplace_kernel<<<(unsigned)((nb + 255) / 256), 256, 0, stream>>>(hB, nb);
    }

    // out = [h0|hA|hB] @ Wc + bc
    cls_gemm_kernel<<<gemm_blocks, 256, BC_BYTES, stream>>>(
        h0, hA, hB, (const char*)Bc, cls_b, out, N);
}
